// SelectiveRecurrentAttention_59030030516255
// MI455X (gfx1250) — matrix operand order repaired and checked by reading, not compiled, not run on hardware
//
#include <hip/hip_runtime.h>
#include <hip/hip_bf16.h>

typedef __attribute__((ext_vector_type(16))) _Float16 v16h;
typedef __attribute__((ext_vector_type(8)))  _Float16 v8h;
typedef __attribute__((ext_vector_type(8)))  float    v8f;

#define B_   2
#define T_   2048
#define DM   512
#define H_   8
#define DH   64
#define QKV_LD 1536
#define TAU_ 20.0f
#define NROW (B_*T_)   // 4096

__device__ __forceinline__ v8f wmma_f16(v16h a, v16h b, v8f c) {
  return __builtin_amdgcn_wmma_f32_16x16x32_f16(false, a, false, b, (short)0, c, false, false);
}
__device__ __forceinline__ v16h cat8(v8h a, v8h b) {
  return __builtin_shufflevector(a, b, 0,1,2,3,4,5,6,7,8,9,10,11,12,13,14,15);
}
// gfx1250 async global->LDS copy (ASYNCcnt); 16B per lane.
__device__ __forceinline__ void async_b128(unsigned int lds_off, unsigned long long gaddr) {
  asm volatile("global_load_async_to_lds_b128 %0, %1, off"
               :: "v"(lds_off), "v"(gaddr) : "memory");
}
__device__ __forceinline__ void wait_async0() {
  asm volatile("s_wait_asynccnt 0x0" ::: "memory");
}

// ---------------------------------------------------------------------------
// elementwise f32 -> f16 (n multiple of 1024)
// ---------------------------------------------------------------------------
__global__ __launch_bounds__(256)
void cvt_f16(const float* __restrict__ s, _Float16* __restrict__ d, int n) {
  int i = (blockIdx.x * 256 + threadIdx.x) * 4;
  if (i >= n) return;
  float4 x = *(const float4*)(s + i);
  union { _Float16 h[4]; unsigned long long u; } o;
  o.h[0] = (_Float16)x.x; o.h[1] = (_Float16)x.y;
  o.h[2] = (_Float16)x.z; o.h[3] = (_Float16)x.w;
  *(unsigned long long*)(d + i) = o.u;
}

// ---------------------------------------------------------------------------
// C[M,N] = act( A[M,K] @ Bw[N,K]^T ), f16 in / f32 out. Block 128 = 4 waves.
// ---------------------------------------------------------------------------
__global__ __launch_bounds__(128)
void gemm_nt_wmma_f16(const _Float16* __restrict__ A, const _Float16* __restrict__ Bw,
                      float* __restrict__ C, int N, int K, int act) {
  const int lane = threadIdx.x & 31;
  const int w    = threadIdx.x >> 5;
  const int l16  = lane & 15;
  const int hi   = lane >> 4;
  const int m0   = blockIdx.y * 64 + w * 16;
  const int n0   = blockIdx.x * 64;

  v8f acc[4] = {};
  const _Float16* arow = A + (size_t)(m0 + l16) * K;
  for (int k0 = 0; k0 < K; k0 += 32) {
    v16h a = *(const v16h*)(arow + k0 + hi*16);
    #pragma unroll
    for (int nt = 0; nt < 4; ++nt) {
      v16h b = *(const v16h*)(Bw + (size_t)(n0 + nt*16 + l16) * K + k0 + hi*16);
      acc[nt] = wmma_f16(a, b, acc[nt]);
    }
  }
  #pragma unroll
  for (int nt = 0; nt < 4; ++nt)
    #pragma unroll
    for (int v = 0; v < 8; ++v) {
      int row = m0 + hi*8 + v;
      int col = n0 + nt*16 + l16;
      float x = acc[nt][v];
      if (act == 1) x = x / (1.0f + __expf(-x));     // SiLU
      C[(size_t)row * N + col] = x;
    }
}

// ---------------------------------------------------------------------------
// xPos rotary on q (x scale, x0.125 retention scale) and k (/ scale),
// f32 qkv -> f16 qh/kh (row-major) and f16 vT (b,h,dv,T transposed).
// ---------------------------------------------------------------------------
__global__ __launch_bounds__(256)
void xpos_split_kernel(const float* __restrict__ qkv, _Float16* __restrict__ qh,
                       _Float16* __restrict__ kh, _Float16* __restrict__ vT) {
  int idx = blockIdx.x * blockDim.x + threadIdx.x;      // NROW * 256
  if (idx >= NROW * (DM/2)) return;
  int p   = idx & (DM/2 - 1);
  int row = idx >> 8;
  int t   = row & (T_ - 1);
  int b   = row >> 11;
  float base  = (2.0f * (float)p + 0.4f * (float)DM) / (1.4f * (float)DM);
  float scale = powf(base, (float)t / 512.0f);          // SCALE_BASE
  float invf  = powf(10000.0f, -(float)p / (float)(DM/2));
  float sn, cs; __sincosf((float)t * invf, &sn, &cs);

  const float* qp = qkv + (size_t)row * QKV_LD + 2*p;
  float q0 = qp[0], q1 = qp[1];
  float cq = cs * scale, sq = sn * scale;
  qh[(size_t)row * DM + 2*p]     = (_Float16)((q0*cq - q1*sq) * 0.125f);
  qh[(size_t)row * DM + 2*p + 1] = (_Float16)((q1*cq + q0*sq) * 0.125f);

  const float* kp = qkv + (size_t)row * QKV_LD + DM + 2*p;
  float k0 = kp[0], k1 = kp[1];
  float ck = cs / scale, sk = sn / scale;
  kh[(size_t)row * DM + 2*p]     = (_Float16)(k0*ck - k1*sk);
  kh[(size_t)row * DM + 2*p + 1] = (_Float16)(k1*ck + k0*sk);

  int h  = p >> 5;
  int d0 = (2*p) & 63;
  const float* vp = qkv + (size_t)row * QKV_LD + 2*DM + 2*p;
  size_t vbase = ((size_t)((b*H_ + h)*DH + d0)) * T_ + t;
  vT[vbase]      = (_Float16)vp[0];
  vT[vbase + T_] = (_Float16)vp[1];
}

// ---------------------------------------------------------------------------
// log-decay logits: dec[row,h] = log(sigmoid(hs@gamma_w^T + b)) / TAU.
// ---------------------------------------------------------------------------
__global__ __launch_bounds__(256)
void decay_logit_kernel(const float* __restrict__ hs, const float* __restrict__ gw,
                        const float* __restrict__ gb, float* __restrict__ dec) {
  int row  = blockIdx.x * 8 + (threadIdx.x >> 5);
  int lane = threadIdx.x & 31;
  if (row >= NROW) return;
  float acc[H_];
  #pragma unroll
  for (int h = 0; h < H_; ++h) acc[h] = 0.0f;
  const float* hrow = hs + (size_t)row * DM;
  for (int i = 0; i < DM/32; ++i) {
    int kk = lane + i*32;
    float x = hrow[kk];
    #pragma unroll
    for (int h = 0; h < H_; ++h) acc[h] += x * gw[h*DM + kk];
  }
  #pragma unroll
  for (int h = 0; h < H_; ++h)
    #pragma unroll
    for (int off = 16; off > 0; off >>= 1)
      acc[h] += __shfl_xor(acc[h], off, 32);
  if (lane == 0) {
    #pragma unroll
    for (int h = 0; h < H_; ++h) {
      float z  = acc[h] + gb[h];
      float ls = (z < 0.0f) ? (z - log1pf(__expf(z))) : (-log1pf(__expf(-z)));
      dec[(size_t)row * H_ + h] = ls / TAU_;
    }
  }
}

// ---------------------------------------------------------------------------
// Inclusive cumsum of log-decay along T for one (b,h) per block.
// ---------------------------------------------------------------------------
__global__ __launch_bounds__(256)
void decay_scan_kernel(const float* __restrict__ dec, float* __restrict__ L) {
  int b = blockIdx.x >> 3, h = blockIdx.x & 7;
  __shared__ float ps[256];
  float vloc[8]; float run = 0.0f;
  #pragma unroll
  for (int it = 0; it < 8; ++it) {
    int t = threadIdx.x * 8 + it;
    run += dec[((size_t)(b*T_ + t)) * H_ + h];
    vloc[it] = run;
  }
  ps[threadIdx.x] = run;
  __syncthreads();
  float total = run;
  for (int off = 1; off < 256; off <<= 1) {
    float add = (threadIdx.x >= off) ? ps[threadIdx.x - off] : 0.0f;
    __syncthreads();
    ps[threadIdx.x] += add;
    __syncthreads();
  }
  float excl = ps[threadIdx.x] - total;
  #pragma unroll
  for (int it = 0; it < 8; ++it) {
    int t = threadIdx.x * 8 + it;
    L[((size_t)(b*T_ + t)) * H_ + h] = excl + vloc[it];
  }
}

// ---------------------------------------------------------------------------
// Flash-style causal retention. Grid (T/64, B*H), block 128 (4 waves).
// K/V tiles staged via GLOBAL_LOAD_ASYNC_TO_LDS_B128; WMMA for S and PV.
// ---------------------------------------------------------------------------
__global__ __launch_bounds__(128)
void retention_kernel(const _Float16* __restrict__ qh, const _Float16* __restrict__ kh,
                      const _Float16* __restrict__ vT, const float* __restrict__ L,
                      float* __restrict__ ret) {
  __shared__ _Float16 Klds[64*64];          // [t_local][d]
  __shared__ _Float16 Vlds[64*64];          // [vdim][t_local]
  __shared__ _Float16 Plds[4][16*64];       // per-wave P strip

  const int bh = blockIdx.y; const int b = bh >> 3, h = bh & 7;
  const int i0 = blockIdx.x * 64;
  const int w  = threadIdx.x >> 5;
  const int lane = threadIdx.x & 31;
  const int l16 = lane & 15, hi = lane >> 4;
  const int ibase = i0 + w * 16;
  const size_t rowbase = (size_t)(b * T_);

  const unsigned int kldsb = (unsigned int)(uintptr_t)&Klds[0];
  const unsigned int vldsb = (unsigned int)(uintptr_t)&Vlds[0];

  // Q A-fragments (pre-scaled f16), loaded once per i-tile.
  const _Float16* qrow = qh + (rowbase + ibase + l16) * DM + h * DH;
  v16h qa[2];
  #pragma unroll
  for (int ks = 0; ks < 2; ++ks)
    qa[ks] = *(const v16h*)(qrow + ks*32 + hi*16);

  float Li[8];
  #pragma unroll
  for (int v = 0; v < 8; ++v)
    Li[v] = L[(rowbase + ibase + hi*8 + v) * H_ + h];

  v8f o[4] = {};

  for (int jt = 0; jt <= (int)blockIdx.x; ++jt) {
    const int j0 = jt * 64;
    __syncthreads();
    // async-stage K (row-major) and V (pre-transposed) 64x64 f16 tiles:
    // 512 16B chunks each; 4 waves x 4 issues x 32 lanes.
    #pragma unroll
    for (int q = 0; q < 4; ++q) {
      int chunk = (w*4 + q)*32 + lane;        // 0..511
      int row = chunk >> 3, c = chunk & 7;
      unsigned long long kg = (unsigned long long)(uintptr_t)
          (kh + (rowbase + j0 + row) * DM + h*DH + c*8);
      async_b128(kldsb + (unsigned int)(row*128 + c*16), kg);
      unsigned long long vg = (unsigned long long)(uintptr_t)
          (vT + ((size_t)(bh*DH + row)) * T_ + j0 + c*8);
      async_b128(vldsb + (unsigned int)(row*128 + c*16), vg);
    }
    wait_async0();
    __syncthreads();

    // S = Q K^T, decay mask, write P strip to per-wave LDS
    #pragma unroll
    for (int nt = 0; nt < 4; ++nt) {
      v8f s = {};
      #pragma unroll
      for (int ks = 0; ks < 2; ++ks) {
        v16h bb = *(const v16h*)&Klds[(nt*16 + l16)*64 + ks*32 + hi*16];
        s = wmma_f16(qa[ks], bb, s);
      }
      const int jcol = j0 + nt*16 + l16;
      float Lj = L[(rowbase + jcol) * H_ + h];
      #pragma unroll
      for (int v = 0; v < 8; ++v) {
        int irow = ibase + hi*8 + v;
        float p = (jcol <= irow) ? s[v] * __expf(Li[v] - Lj) : 0.0f;
        Plds[w][(hi*8 + v)*64 + nt*16 + l16] = (_Float16)p;
      }
    }

    // O += P @ V
    v16h pa[2];
    #pragma unroll
    for (int ks = 0; ks < 2; ++ks) {
      const _Float16* pp = &Plds[w][l16*64 + ks*32];
      pa[ks] = *(const v16h*)(pp + hi*16);
    }
    #pragma unroll
    for (int nt = 0; nt < 4; ++nt)
      #pragma unroll
      for (int ks = 0; ks < 2; ++ks) {
        v16h bb = *(const v16h*)&Vlds[(nt*16 + l16)*64 + ks*32 + hi*16];
        o[nt] = wmma_f16(pa[ks], bb, o[nt]);
      }
  }

  #pragma unroll
  for (int nt = 0; nt < 4; ++nt)
    #pragma unroll
    for (int v = 0; v < 8; ++v) {
      int irow = ibase + hi*8 + v;
      int col  = h*DH + nt*16 + l16;
      ret[(rowbase + irow) * DM + col] = o[nt][v];
    }
}

// ---------------------------------------------------------------------------
// Per-head groupnorm of retention output * SiLU gate -> yh (f16).
// ---------------------------------------------------------------------------
__global__ __launch_bounds__(256)
void gnorm_gate_kernel(const float* __restrict__ ret, const float* __restrict__ gate,
                       _Float16* __restrict__ yh) {
  int g = blockIdx.x * 8 + (threadIdx.x >> 5);
  int lane = threadIdx.x & 31;
  if (g >= NROW * H_) return;
  int row = g >> 3, h = g & 7;
  const float* base = ret + (size_t)row * DM + h * DH;
  float e0 = base[lane], e1 = base[lane + 32];
  float s = e0 + e1, sq = e0*e0 + e1*e1;
  #pragma unroll
  for (int off = 16; off > 0; off >>= 1) {
    s  += __shfl_xor(s,  off, 32);
    sq += __shfl_xor(sq, off, 32);
  }
  float mean = s * (1.0f/64.0f);
  float var  = sq * (1.0f/64.0f) - mean*mean;
  float inv  = rsqrtf(var + 1e-5f);
  const float* gb = gate + (size_t)row * DM + h * DH;
  _Float16* yb = yh + (size_t)row * DM + h * DH;
  yb[lane]      = (_Float16)((e0 - mean) * inv * gb[lane]);
  yb[lane + 32] = (_Float16)((e1 - mean) * inv * gb[lane + 32]);
}

// ---------------------------------------------------------------------------
// curr_kv[b,h,dk,dv] = sum_t k[t,dk] * v[t,dv] * exp(L[T-1]-L[t]).
// ---------------------------------------------------------------------------
__global__ __launch_bounds__(256)
void currkv_kernel(const _Float16* __restrict__ kh, const float* __restrict__ qkv,
                   const float* __restrict__ L, float* __restrict__ ckv) {
  int bh = blockIdx.x; int b = bh >> 3, h = bh & 7;
  int dk  = threadIdx.x >> 2;
  int dv0 = (threadIdx.x & 3) * 16;
  float acc[16] = {};
  size_t rowbase = (size_t)(b * T_);
  float Llast = L[(rowbase + T_ - 1) * H_ + h];
  for (int t = 0; t < T_; ++t) {
    float wgt = __expf(Llast - L[(rowbase + t) * H_ + h]);
    float kd = (float)kh[(rowbase + t) * DM + h*DH + dk] * wgt;
    const float* vr = qkv + (rowbase + t) * QKV_LD + 2*DM + h*DH + dv0;
    #pragma unroll
    for (int j = 0; j < 16; ++j) acc[j] += kd * vr[j];
  }
  float* out = ckv + ((size_t)(b*H_ + h) * DH + dk) * DH + dv0;
  #pragma unroll
  for (int j = 0; j < 16; ++j) out[j] = acc[j];
}

// ---------------------------------------------------------------------------
extern "C" void kernel_launch(void* const* d_in, const int* in_sizes, int n_in,
                              void* d_out, int out_size, void* d_ws, size_t ws_size,
                              hipStream_t stream) {
  (void)in_sizes; (void)n_in; (void)out_size; (void)ws_size;
  const float* hs      = (const float*)d_in[0];
  /* d_in[1] = t (unused by reference) */
  const float* qkv_w   = (const float*)d_in[2];
  const float* gamma_w = (const float*)d_in[3];
  const float* gamma_b = (const float*)d_in[4];
  const float* gated_w = (const float*)d_in[5];
  const float* out_w   = (const float*)d_in[6];

  float* out = (float*)d_out;                       // (B,T,512)
  float* ckv = out + (size_t)NROW * DM;             // (B,H,64,64)

  char* p = (char*)d_ws;
  float*    qkv  = (float*)p;    p += (size_t)NROW*QKV_LD*4;
  _Float16* hsh  = (_Float16*)p; p += (size_t)NROW*DM*2;
  _Float16* wqh  = (_Float16*)p; p += (size_t)QKV_LD*DM*2;
  _Float16* wgh  = (_Float16*)p; p += (size_t)DM*DM*2;
  _Float16* woh  = (_Float16*)p; p += (size_t)DM*DM*2;
  _Float16* qh   = (_Float16*)p; p += (size_t)NROW*DM*2;
  _Float16* khb  = (_Float16*)p; p += (size_t)NROW*DM*2;
  _Float16* vTb  = (_Float16*)p; p += (size_t)NROW*DM*2;
  float*    dec  = (float*)p;    p += (size_t)NROW*H_*4;
  float*    Lbuf = (float*)p;    p += (size_t)NROW*H_*4;
  float*    ret  = (float*)p;    p += (size_t)NROW*DM*4;
  float*    gate = (float*)p;    p += (size_t)NROW*DM*4;
  _Float16* yh   = (_Float16*)p; p += (size_t)NROW*DM*2;

  cvt_f16<<<(NROW*DM)/1024, 256, 0, stream>>>(hs, hsh, NROW*DM);
  cvt_f16<<<(QKV_LD*DM)/1024, 256, 0, stream>>>(qkv_w, wqh, QKV_LD*DM);
  cvt_f16<<<(DM*DM)/1024, 256, 0, stream>>>(gated_w, wgh, DM*DM);
  cvt_f16<<<(DM*DM)/1024, 256, 0, stream>>>(out_w, woh, DM*DM);
  gemm_nt_wmma_f16<<<dim3(QKV_LD/64, NROW/64), 128, 0, stream>>>(hsh, wqh, qkv, QKV_LD, DM, 0);
  xpos_split_kernel<<<(NROW*(DM/2))/256, 256, 0, stream>>>(qkv, qh, khb, vTb);
  decay_logit_kernel<<<NROW/8, 256, 0, stream>>>(hs, gamma_w, gamma_b, dec);
  decay_scan_kernel<<<B_*H_, 256, 0, stream>>>(dec, Lbuf);
  retention_kernel<<<dim3(T_/64, B_*H_), 128, 0, stream>>>(qh, khb, vTb, Lbuf, ret);
  gemm_nt_wmma_f16<<<dim3(DM/64, NROW/64), 128, 0, stream>>>(hsh, wgh, gate, DM, DM, 1);
  gnorm_gate_kernel<<<(NROW*H_)/8, 256, 0, stream>>>(ret, gate, yh);
  gemm_nt_wmma_f16<<<dim3(DM/64, NROW/64), 128, 0, stream>>>(yh, woh, out, DM, DM, 0);
  currkv_kernel<<<B_*H_, 256, 0, stream>>>(khb, qkv, Lbuf, ckv);
}
